// Model_43044162240935
// MI455X (gfx1250) — matrix operand order repaired and checked by reading, not compiled, not run on hardware
//
#include <hip/hip_runtime.h>
#include <hip/hip_bf16.h>

// ---------------------------------------------------------------------------
// CDNA5 (gfx1250) GCN pipeline: bf16 WMMA dense GEMMs (pre-packed B operand)
// + native-atomic SpMM scatter.
// ---------------------------------------------------------------------------

typedef __attribute__((ext_vector_type(16))) __bf16 v16bf;
typedef __attribute__((ext_vector_type(8)))  float  v8f;

// ---------------------------------------------------------------------------
// Weight packer: W[K,Nout] fp32 (row-major) -> bf16 tiles in the contiguous-k WMMA
// B-operand register layout. One 32x16 (KxN) tile = 32 lanes x v16bf:
//   lanes 0-15 : col n=lane,    K = kt*32 + 0..15
//   lanes 16-31: col n=lane-16, K = kt*32 + 16..31
// Tile order: tile = kt * (Nout/16) + nt. Element i of a lane = ascending K.
// ---------------------------------------------------------------------------
__global__ void __launch_bounds__(256)
pack_w_bf16(const float* __restrict__ W, __bf16* __restrict__ Wp,
            int K, int Nout) {
    const int idx = blockIdx.x * blockDim.x + threadIdx.x;
    const int total = K * Nout;
    if (idx >= total) return;
    const int tiles16 = Nout >> 4;
    const int tile = idx >> 9;          // 512 bf16 per tile
    const int r    = idx & 511;
    const int lane = r >> 4;
    const int i    = r & 15;
    const int kt = tile / tiles16;
    const int nt = tile - kt * tiles16;
    const int half = lane >> 4;
    const int n    = lane & 15;
    const int k = kt * 32 + 16 * half + i;
    Wp[idx] = (__bf16)W[(size_t)k * Nout + nt * 16 + n];
}

// ---------------------------------------------------------------------------
// Dense GEMM: Y[M,Nout] = X[M,K] @ W[K,Nout], fp32 X/Y, B pre-packed bf16.
// One wave computes a 16x64 output tile; K stepped by 32.
// Requires: M % 16 == 0, K % 32 == 0, Nout % 64 == 0.
// ---------------------------------------------------------------------------
__global__ void __launch_bounds__(128)
gemm_wmma_bf16(const float* __restrict__ X, const v16bf* __restrict__ Wp,
               float* __restrict__ Y, int M, int K, int Nout) {
    const int wavesPerBlock = blockDim.x >> 5;
    const int wave = blockIdx.x * wavesPerBlock + (threadIdx.x >> 5);
    const int lane = threadIdx.x & 31;
    const int tilesN = Nout >> 6;       // 64-wide wave tiles
    const int tilesM = M >> 4;
    if (wave >= tilesM * tilesN) return;
    const int tm = wave / tilesN;
    const int tn = (wave % tilesN) << 6;
    const int tiles16 = Nout >> 4;
    const int nt0 = tn >> 4;            // first 16-col tile index
    const int half = lane >> 4;
    const int m16  = lane & 15;

    const float* __restrict__ xrow = X + (size_t)(tm * 16 + m16) * K;

    v8f c0 = {}, c1 = {}, c2 = {}, c3 = {};

    for (int kk = 0; kk < K; kk += 32) {
        // ---- A tile (16x32 bf16), contiguous k order, same as packed B:
        // lanes 0-15 : row m16, K = kk+{0..15}
        // lanes 16-31: row m16, K = kk+{16..31}
        const int ka = kk + 16 * half;
        const int kb = kk + 16 * half + 8;
        const float4 f0 = *(const float4*)(xrow + ka);
        const float4 f1 = *(const float4*)(xrow + ka + 4);
        const float4 f2 = *(const float4*)(xrow + kb);
        const float4 f3 = *(const float4*)(xrow + kb + 4);
        v16bf a;
        a[0]  = (__bf16)f0.x; a[1]  = (__bf16)f0.y;
        a[2]  = (__bf16)f0.z; a[3]  = (__bf16)f0.w;
        a[4]  = (__bf16)f1.x; a[5]  = (__bf16)f1.y;
        a[6]  = (__bf16)f1.z; a[7]  = (__bf16)f1.w;
        a[8]  = (__bf16)f2.x; a[9]  = (__bf16)f2.y;
        a[10] = (__bf16)f2.z; a[11] = (__bf16)f2.w;
        a[12] = (__bf16)f3.x; a[13] = (__bf16)f3.y;
        a[14] = (__bf16)f3.z; a[15] = (__bf16)f3.w;

        // ---- B tiles: one contiguous 32-byte v16bf per lane per tile.
        const size_t btile = (size_t)(kk >> 5) * tiles16 + nt0;
        const v16bf b0 = Wp[(btile + 0) * 32 + lane];
        const v16bf b1 = Wp[(btile + 1) * 32 + lane];
        const v16bf b2 = Wp[(btile + 2) * 32 + lane];
        const v16bf b3 = Wp[(btile + 3) * 32 + lane];

        c0 = __builtin_amdgcn_wmma_f32_16x16x32_bf16(false, a, false, b0,
                                                     (short)0, c0, false, false);
        c1 = __builtin_amdgcn_wmma_f32_16x16x32_bf16(false, a, false, b1,
                                                     (short)0, c1, false, false);
        c2 = __builtin_amdgcn_wmma_f32_16x16x32_bf16(false, a, false, b2,
                                                     (short)0, c2, false, false);
        c3 = __builtin_amdgcn_wmma_f32_16x16x32_bf16(false, a, false, b3,
                                                     (short)0, c3, false, false);
    }

    // ---- Store. f32 C/D layout: VGPR r -> row r (lanes 0-15) / row r+8.
    const int rowBase = tm * 16 + 8 * half;
    const int colBase = tn + m16;
#pragma unroll
    for (int r = 0; r < 8; ++r) {
        float* __restrict__ yr = Y + (size_t)(rowBase + r) * Nout + colBase;
        yr[0]  = c0[r];
        yr[16] = c1[r];
        yr[32] = c2[r];
        yr[48] = c3[r];
    }
}

// ---------------------------------------------------------------------------
// SpMM scatter: Y[row[e], :] += val[e] * X[col[e], :]  (Y pre-zeroed)
// One thread per (edge, 4-feature chunk); native f32 atomics into L2.
// ---------------------------------------------------------------------------
__global__ void __launch_bounds__(256)
spmm_atomic(const int* __restrict__ row, const int* __restrict__ col,
            const float* __restrict__ val, const float* __restrict__ X,
            float* __restrict__ Y, int E, int dim) {
    const int chunks = dim >> 2;
    const long long idx = (long long)blockIdx.x * blockDim.x + threadIdx.x;
    const long long total = (long long)E * chunks;
    if (idx >= total) return;
    const int e = (int)(idx / chunks);
    const int c = (int)(idx % chunks) << 2;
    const int s = col[e];
    const int d = row[e];
    const float v = val[e];
    const float4 xv = *(const float4*)(X + (size_t)s * dim + c);
    float* __restrict__ yp = Y + (size_t)d * dim + c;
    unsafeAtomicAdd(yp + 0, v * xv.x);
    unsafeAtomicAdd(yp + 1, v * xv.y);
    unsafeAtomicAdd(yp + 2, v * xv.z);
    unsafeAtomicAdd(yp + 3, v * xv.w);
}

// ---------------------------------------------------------------------------
__global__ void __launch_bounds__(256)
zero_f32_v4(float4* __restrict__ p, long long n4) {
    long long i = (long long)blockIdx.x * blockDim.x + threadIdx.x;
    if (i < n4) p[i] = make_float4(0.f, 0.f, 0.f, 0.f);
}

// y = relu(prelu(y + b[c], a))   (layer-0 epilogue), float4-vectorized
__global__ void __launch_bounds__(256)
bias_prelu_relu_v4(float4* __restrict__ y, const float* __restrict__ b,
                   const float* __restrict__ a, long long n4, int dim) {
    long long i = (long long)blockIdx.x * blockDim.x + threadIdx.x;
    if (i >= n4) return;
    const int c = (int)((i << 2) % dim);
    const float4 bv = *(const float4*)(b + c);
    const float alpha = a[0];
    float4 z = y[i];
    z.x += bv.x; z.y += bv.y; z.z += bv.z; z.w += bv.w;
    z.x = fmaxf((z.x >= 0.f) ? z.x : alpha * z.x, 0.f);
    z.y = fmaxf((z.y >= 0.f) ? z.y : alpha * z.y, 0.f);
    z.z = fmaxf((z.z >= 0.f) ? z.z : alpha * z.z, 0.f);
    z.w = fmaxf((z.w >= 0.f) ? z.w : alpha * z.w, 0.f);
    y[i] = z;
}

// y = prelu(y + b[c], a)   (layer-1 epilogue), float4-vectorized
__global__ void __launch_bounds__(256)
bias_prelu_v4(float4* __restrict__ y, const float* __restrict__ b,
              const float* __restrict__ a, long long n4, int dim) {
    long long i = (long long)blockIdx.x * blockDim.x + threadIdx.x;
    if (i >= n4) return;
    const int c = (int)((i << 2) % dim);
    const float4 bv = *(const float4*)(b + c);
    const float alpha = a[0];
    float4 z = y[i];
    z.x += bv.x; z.y += bv.y; z.z += bv.z; z.w += bv.w;
    z.x = (z.x >= 0.f) ? z.x : alpha * z.x;
    z.y = (z.y >= 0.f) ? z.y : alpha * z.y;
    z.z = (z.z >= 0.f) ? z.z : alpha * z.z;
    z.w = (z.w >= 0.f) ? z.w : alpha * z.w;
    y[i] = z;
}

// ---------------------------------------------------------------------------
// Gate: beta = sigmoid([h1@G1w+G1b, h3@G2w+G2b, deg] @ G3w + G3b)
// One wave per node; lanes split the 64 gate dims, shfl_xor reduction.
// ---------------------------------------------------------------------------
__global__ void __launch_bounds__(256)
gate_beta(const float* __restrict__ h1, const float* __restrict__ h3,
          const float* __restrict__ deg,
          const float* __restrict__ G1w, const float* __restrict__ G1b,
          const float* __restrict__ G2w, const float* __restrict__ G2b,
          const float* __restrict__ G3w, const float* __restrict__ G3b,
          float* __restrict__ beta, int n) {
    const int wavesPerBlock = blockDim.x >> 5;
    const int node = blockIdx.x * wavesPerBlock + (threadIdx.x >> 5);
    const int lane = threadIdx.x & 31;
    if (node >= n) return;
    const float* __restrict__ r1 = h1 + (size_t)node * 128;
    const float* __restrict__ r3 = h3 + (size_t)node * 128;
    float s = 0.0f;
#pragma unroll
    for (int jj = 0; jj < 64; jj += 32) {
        const int j = jj + lane;
        float z1 = G1b[j];
        float z2 = G2b[j];
        for (int k = 0; k < 128; ++k) {
            z1 = fmaf(r1[k], G1w[k * 64 + j], z1);
            z2 = fmaf(r3[k], G2w[k * 64 + j], z2);
        }
        s = fmaf(z1, G3w[j], s);
        s = fmaf(z2, G3w[64 + j], s);
    }
#pragma unroll
    for (int off = 16; off > 0; off >>= 1) s += __shfl_xor(s, off, 32);
    if (lane == 0) {
        s += deg[node] * G3w[128] + G3b[0];
        beta[node] = 1.0f / (1.0f + __expf(-s));
    }
}

// ---------------------------------------------------------------------------
extern "C" void kernel_launch(void* const* d_in, const int* in_sizes, int n_in,
                              void* d_out, int out_size, void* d_ws, size_t ws_size,
                              hipStream_t stream) {
    (void)n_in; (void)out_size; (void)ws_size;
    const int N = 50000, IN = 256, H2 = 256, H = 128;

    const float* x1  = (const float*)d_in[0];
    const float* x2  = (const float*)d_in[1];
    const int*   A1r = (const int*)  d_in[2];
    const int*   A1c = (const int*)  d_in[3];
    const float* A1v = (const float*)d_in[4];
    const int*   A2r = (const int*)  d_in[5];
    const int*   A2c = (const int*)  d_in[6];
    const float* A2v = (const float*)d_in[7];
    const int*   Dr  = (const int*)  d_in[8];
    const int*   Dc  = (const int*)  d_in[9];
    const float* Dv  = (const float*)d_in[10];
    const float* deg = (const float*)d_in[11];
    const float* W0  = (const float*)d_in[12];
    const float* b0  = (const float*)d_in[13];
    const float* a0  = (const float*)d_in[14];
    const float* W1  = (const float*)d_in[15];
    const float* b1  = (const float*)d_in[16];
    const float* a1p = (const float*)d_in[17];
    const float* G1w = (const float*)d_in[18];
    const float* G1b = (const float*)d_in[19];
    const float* G2w = (const float*)d_in[20];
    const float* G2b = (const float*)d_in[21];
    const float* G3w = (const float*)d_in[22];
    const float* G3b = (const float*)d_in[23];

    const int E  = in_sizes[2];   // 800000
    const int ED = in_sizes[8];   // 400000

    // Workspace: XW (N x 256) | T0 (N x 256) | G (N x 128) | packed weights
    float* XW = (float*)d_ws;
    float* T0 = XW + (size_t)N * H2;
    float* G  = T0 + (size_t)N * H2;
    __bf16* W0p = (__bf16*)(G + (size_t)N * H);      // 256*256 bf16
    __bf16* W1p = W0p + (size_t)IN * H2;             // 256*128 bf16

    float* out  = (float*)d_out;
    float* h1   = out;
    float* h2   = out + 1 * (size_t)N * H;
    float* h3   = out + 2 * (size_t)N * H;
    float* h4   = out + 3 * (size_t)N * H;
    float* beta = out + 4 * (size_t)N * H;

    auto gemm = [&](const float* X, const __bf16* Wp, float* Y,
                    int M, int K, int Nout) {
        const int tiles  = (M / 16) * (Nout / 64);
        const int blocks = (tiles + 3) / 4;            // 4 waves / 128-thread block
        gemm_wmma_bf16<<<blocks, 128, 0, stream>>>(X, (const v16bf*)Wp, Y,
                                                   M, K, Nout);
    };
    auto zero = [&](float* p, long long n) {
        const long long n4 = n >> 2;
        zero_f32_v4<<<(int)((n4 + 255) / 256), 256, 0, stream>>>((float4*)p, n4);
    };
    auto spmm = [&](const int* r, const int* c, const float* v,
                    const float* X, float* Y, int e, int dim) {
        const long long total = (long long)e * (dim / 4);
        spmm_atomic<<<(int)((total + 255) / 256), 256, 0, stream>>>(r, c, v, X, Y,
                                                                    e, dim);
    };
    auto branch = [&](const int* r, const int* c, const float* v, int e,
                      float* hout) {
        const long long n0 = (long long)N * H2, n04 = n0 >> 2;
        const long long n1 = (long long)N * H,  n14 = n1 >> 2;
        zero(T0, n0);
        spmm(r, c, v, XW, T0, e, H2);
        bias_prelu_relu_v4<<<(int)((n04 + 255) / 256), 256, 0, stream>>>(
            (float4*)T0, b0, a0, n04, H2);
        gemm(T0, W1p, G, N, H2, H);
        zero(hout, n1);
        spmm(r, c, v, G, hout, e, H);
        bias_prelu_v4<<<(int)((n14 + 255) / 256), 256, 0, stream>>>(
            (float4*)hout, b1, a1p, n14, H);
    };

    // Pack both weight matrices into WMMA B-operand layout (tiny, per launch).
    pack_w_bf16<<<(IN * H2 + 255) / 256, 256, 0, stream>>>(W0, W0p, IN, H2);
    pack_w_bf16<<<(H2 * H + 255) / 256, 256, 0, stream>>>(W1, W1p, H2, H);

    // Branches sharing x1@W0 (h1, h3, h4), then x2@W0 (h2).
    gemm(x1, W0p, XW, N, IN, H2);
    branch(A1r, A1c, A1v, E,  h1);
    branch(A2r, A2c, A2v, E,  h3);
    branch(Dr,  Dc,  Dv,  ED, h4);
    gemm(x2, W0p, XW, N, IN, H2);
    branch(A1r, A1c, A1v, E,  h2);

    // Gate / beta (wave per node, 8 waves per 256-thread block).
    gate_beta<<<(N + 7) / 8, 256, 0, stream>>>(h1, h3, deg, G1w, G1b,
                                               G2w, G2b, G3w, G3b, beta, N);
}
